// SequentialTransductionUnitJagged_34849364639830
// MI455X (gfx1250) — matrix operand order repaired and checked by reading, not compiled, not run on hardware
//
#include <hip/hip_runtime.h>

// ---------------------------------------------------------------------------
// HSTU jagged attention block for MI455X (gfx1250, wave32, WMMA bf16).
// B=8, N=1024, D=512, H=8, DA=DV=64, T=8192, UVQK cols=2048.
// Hot loops: per-lane base pointers + immediate offsets. GEMM K-steps are
// pipelined with sched_group_barrier: [18 x VMEM-read][8 x WMMA] per step so
// fragment loads of step i+1 overlap the matrix pipe of step i.
// ---------------------------------------------------------------------------

typedef __attribute__((ext_vector_type(16))) __bf16 bf16x16;
typedef __attribute__((ext_vector_type(8)))  __bf16 bf16x8;
typedef __attribute__((ext_vector_type(8)))  float  f32x8;
typedef __attribute__((ext_vector_type(4)))  float  f32x4;

union BF16x16U { bf16x16 v; bf16x8 h[2]; };

#define WMMA_BF16(a, b, c) \
  __builtin_amdgcn_wmma_f32_16x16x32_bf16(false, (a), false, (b), (short)0, (c), false, false)

// sched_group_barrier masks
#define SG_MFMA 0x008   // MFMA / WMMA
#define SG_VMEMR 0x020  // VMEM read

__device__ __forceinline__ float silu_f(float x) {
  return x / (1.0f + __expf(-x));
}

// A fragment (16x32, row-major, ld elems): lane base = A + (row0 + r)*ld + hi*8.
__device__ __forceinline__ bf16x16 frag_a_at(const __bf16* __restrict__ p, int eoff) {
  BF16x16U u;
  u.h[0] = *(const bf16x8*)(p + eoff);
  u.h[1] = *(const bf16x8*)(p + eoff + 16);
  return u.v;
}

// B fragment (32x16 KxN, column-major, ld elems): lane base = B + (col0 + r)*ld + hi*8.
__device__ __forceinline__ bf16x16 frag_b_at(const __bf16* __restrict__ p, int eoff) {
  BF16x16U u;
  u.h[0] = *(const bf16x8*)(p + eoff);
  u.h[1] = *(const bf16x8*)(p + eoff + 16);
  return u.v;
}

// ---------------------------------------------------------------------------
// Kernel 1: LayerNorm(x) over D=512, emit bf16. One wave per row.
// ---------------------------------------------------------------------------
__global__ void k_ln_x(const float* __restrict__ x, __bf16* __restrict__ xn) {
  const int row  = blockIdx.x * 8 + (threadIdx.x >> 5);
  const int lane = threadIdx.x & 31;
  const float* xr = x + (size_t)row * 512 + lane * 16;
  f32x4 a[4];
  a[0] = *(const f32x4*)(xr + 0);
  a[1] = *(const f32x4*)(xr + 4);
  a[2] = *(const f32x4*)(xr + 8);
  a[3] = *(const f32x4*)(xr + 12);
  float s = 0.f, sq = 0.f;
#pragma unroll
  for (int i = 0; i < 16; ++i) { float v = a[i >> 2][i & 3]; s += v; sq += v * v; }
#pragma unroll
  for (int o = 16; o > 0; o >>= 1) { s += __shfl_xor(s, o); sq += __shfl_xor(sq, o); }
  const float mean = s * (1.f / 512.f);
  const float var  = sq * (1.f / 512.f) - mean * mean;
  const float rstd = rsqrtf(var + 1e-6f);
  bf16x8 o0, o1;
#pragma unroll
  for (int i = 0; i < 16; ++i) {
    float y = (a[i >> 2][i & 3] - mean) * rstd;
    if (i < 8) o0[i] = (__bf16)y; else o1[i - 8] = (__bf16)y;
  }
  __bf16* dst = xn + (size_t)row * 512 + lane * 16;
  *(bf16x8*)(dst)     = o0;
  *(bf16x8*)(dst + 8) = o1;
}

// ---------------------------------------------------------------------------
// Kernel 2: fp32 (rows x cols, row-major) -> bf16 transposed (cols x rows).
// ---------------------------------------------------------------------------
__global__ void k_transcvt(const float* __restrict__ src, __bf16* __restrict__ dst,
                           int rows, int cols) {
  int idx = blockIdx.x * blockDim.x + threadIdx.x;
  if (idx >= rows * cols) return;
  int kk = idx / cols, c = idx - kk * cols;      // coalesced read
  dst[(size_t)c * rows + kk] = (__bf16)src[idx];
}

// ---------------------------------------------------------------------------
// Kernel 3: GEMM1  mm = silu(xn @ uvqk) -> u(f32), q/k(bf16 row-major),
// v transposed per (b,h). Wave tile 16x128, K fully unrolled (128 WMMAs),
// [18 VMEM][8 WMMA] pipeline groups per K step.
// ---------------------------------------------------------------------------
__global__ void __launch_bounds__(256)
k_gemm1(const __bf16* __restrict__ xn, const __bf16* __restrict__ wT,
        float* __restrict__ u_out, __bf16* __restrict__ q_out,
        __bf16* __restrict__ k_out, __bf16* __restrict__ vT_out) {
  const int wid  = blockIdx.x * 8 + (threadIdx.x >> 5);
  const int mt   = wid & 511;        // 512 M tiles of 16 rows
  const int ns   = wid >> 9;         // 16 N strips of 128 cols
  const int lane = threadIdx.x & 31;
  const int r = lane & 15, hi = lane >> 4;

  const __bf16* pa = xn + (size_t)(mt * 16 + r) * 512 + hi * 8;
  const __bf16* pb = wT + (size_t)(ns * 128 + r) * 512 + hi * 8;

  f32x8 acc[8] = {};
#pragma unroll
  for (int k0 = 0; k0 < 512; k0 += 32) {
    bf16x16 a = frag_a_at(pa, k0);
    bf16x16 bfr[8];
#pragma unroll
    for (int j = 0; j < 8; ++j) bfr[j] = frag_b_at(pb, j * 8192 + k0);  // 16*512
#pragma unroll
    for (int j = 0; j < 8; ++j) acc[j] = WMMA_BF16(a, bfr[j], acc[j]);
    // pipeline: all 18 fragment loads of this step, then its 8 WMMAs
    __builtin_amdgcn_sched_group_barrier(SG_VMEMR, 18, 0);
    __builtin_amdgcn_sched_group_barrier(SG_MFMA,  8,  0);
  }

#pragma unroll
  for (int j = 0; j < 8; ++j) {
    const int c    = ns * 128 + j * 16 + r;   // global column (uniform range per tile)
    const int rowb = mt * 16 + hi * 8;
    if (c < 512) {                                       // u (fp32)
      float* pu = u_out + (size_t)rowb * 512 + c;
#pragma unroll
      for (int e = 0; e < 8; ++e) pu[e * 512] = silu_f(acc[j][e]);
    } else if (c < 1024) {                               // v -> vT[(b*8+h)*64+d][pos]
      const int hh = (c - 512) >> 6, d = (c - 512) & 63;
      const int bb = rowb >> 10, pos = rowb & 1023;
      bf16x8 pk;
#pragma unroll
      for (int e = 0; e < 8; ++e) pk[e] = (__bf16)silu_f(acc[j][e]);
      *(bf16x8*)(vT_out + ((size_t)(bb * 8 + hh) * 64 + d) * 1024 + pos) = pk;
    } else if (c < 1536) {                               // q (bf16 row-major)
      __bf16* pq = q_out + (size_t)rowb * 512 + (c - 1024);
#pragma unroll
      for (int e = 0; e < 8; ++e) pq[e * 512] = (__bf16)silu_f(acc[j][e]);
    } else {                                             // k (bf16 row-major)
      __bf16* pk2 = k_out + (size_t)rowb * 512 + (c - 1536);
#pragma unroll
      for (int e = 0; e < 8; ++e) pk2[e * 512] = (__bf16)silu_f(acc[j][e]);
    }
  }
}

// ---------------------------------------------------------------------------
// Kernel 4: causal attention per (b, h). One wave owns 16 query rows; Q in
// registers for the whole key loop. V fragments prefetched before the S
// epilogue; S goes through a wave-private LDS tile (C->A layout transpose).
// Upper-triangle key blocks skipped; mask loaded only for diagonal tiles.
// ---------------------------------------------------------------------------
__global__ void __launch_bounds__(256)
k_attn(const __bf16* __restrict__ qb, const __bf16* __restrict__ kb,
       const __bf16* __restrict__ vT, const float* __restrict__ mask,
       float* __restrict__ attn) {
  __shared__ __bf16 stile[8][16 * 32];
  const int w    = threadIdx.x >> 5;
  const int wid  = blockIdx.x * 8 + w;
  const int qt   = wid & 63;          // 64 query tiles of 16 rows
  const int h    = (wid >> 6) & 7;
  const int b    = wid >> 9;
  const int lane = threadIdx.x & 31;
  const int r = lane & 15, hi = lane >> 4;

  // per-lane streaming pointers (advance once per 32-key step)
  const __bf16* pk = kb + (size_t)(b * 1024) * 512 + h * 64 + (size_t)r * 512 + hi * 8;
  const __bf16* pv = vT + (size_t)(b * 8 + h) * 64 * 1024 + (size_t)r * 1024 + hi * 8;
  const float*  pm = mask + ((size_t)b * 1024 + qt * 16 + hi * 8) * 1024 + r;

  __bf16* spst = &stile[w][0] + hi * 256 + r;          // store: row=hi*8+e, col=kt*16
  const __bf16* sld = &stile[w][0] + r * 32 + hi * 8;  // A-frag load base (ld=32)

  const __bf16* pq = qb + (size_t)(b * 1024) * 512 + h * 64
                   + (size_t)(qt * 16 + r) * 512 + hi * 8;
  const bf16x16 aq0 = frag_a_at(pq, 0);
  const bf16x16 aq1 = frag_a_at(pq, 32);

  f32x8 oacc[4] = {};
  const int kmax = qt * 16 + 16;                 // causal: keys <= row
  for (int kt0 = 0; kt0 < kmax; kt0 += 32) {
    // prefetch all K fragments (4) and V fragments (4) for this 32-key step
    bf16x16 kf[4], vf[4];
#pragma unroll
    for (int kt = 0; kt < 2; ++kt) {
      kf[kt * 2 + 0] = frag_b_at(pk, kt * 8192 + 0);   // 16*512
      kf[kt * 2 + 1] = frag_b_at(pk, kt * 8192 + 32);
    }
#pragma unroll
    for (int jt = 0; jt < 4; ++jt) vf[jt] = frag_b_at(pv, jt * 16384);  // 16*1024

#pragma unroll
    for (int kt = 0; kt < 2; ++kt) {
      const int colb = kt0 + kt * 16;
      f32x8 s = {};
      s = WMMA_BF16(aq0, kf[kt * 2 + 0], s);
      s = WMMA_BF16(aq1, kf[kt * 2 + 1], s);
      const bool needm = (colb + 15) > (qt * 16);   // tile touches/crosses diagonal
#pragma unroll
      for (int e = 0; e < 8; ++e) {
        float xv = silu_f(s[e]) * (1.0f / 1024.0f);
        if (needm) xv *= pm[(size_t)e * 1024 + colb];
        spst[e * 32 + kt * 16] = (__bf16)xv;
      }
    }
    asm volatile("s_wait_dscnt 0x0" ::: "memory");   // all 32 lanes' S stores visible
    bf16x16 sa;
    {
      BF16x16U u2;
      u2.h[0] = *(const bf16x8*)(sld);
      u2.h[1] = *(const bf16x8*)(sld + 16);
      sa = u2.v;
    }
#pragma unroll
    for (int jt = 0; jt < 4; ++jt) oacc[jt] = WMMA_BF16(sa, vf[jt], oacc[jt]);
    pk += 32 * 512;
    pv += 32;
  }

#pragma unroll
  for (int jt = 0; jt < 4; ++jt) {
    const int col  = h * 64 + jt * 16 + r;
    const int rowb = b * 1024 + qt * 16 + hi * 8;
    float* po = attn + (size_t)rowb * 512 + col;
#pragma unroll
    for (int e = 0; e < 8; ++e) po[e * 512] = oacc[jt][e];
  }
}

// ---------------------------------------------------------------------------
// Kernel 5: o_input = u * LayerNorm(attn_out) over 512, emit bf16.
// ---------------------------------------------------------------------------
__global__ void k_ln_gate(const float* __restrict__ attn, const float* __restrict__ u,
                          __bf16* __restrict__ oin) {
  const int row  = blockIdx.x * 8 + (threadIdx.x >> 5);
  const int lane = threadIdx.x & 31;
  const float* ar = attn + (size_t)row * 512 + lane * 16;
  const float* ur = u    + (size_t)row * 512 + lane * 16;
  f32x4 a[4], g[4];
#pragma unroll
  for (int i = 0; i < 4; ++i) {
    a[i] = *(const f32x4*)(ar + 4 * i);
    g[i] = *(const f32x4*)(ur + 4 * i);
  }
  float s = 0.f, sq = 0.f;
#pragma unroll
  for (int i = 0; i < 16; ++i) { float v = a[i >> 2][i & 3]; s += v; sq += v * v; }
#pragma unroll
  for (int o = 16; o > 0; o >>= 1) { s += __shfl_xor(s, o); sq += __shfl_xor(sq, o); }
  const float mean = s * (1.f / 512.f);
  const float var  = sq * (1.f / 512.f) - mean * mean;
  const float rstd = rsqrtf(var + 1e-6f);
  bf16x8 o0, o1;
#pragma unroll
  for (int i = 0; i < 16; ++i) {
    float y = (a[i >> 2][i & 3] - mean) * rstd * g[i >> 2][i & 3];
    if (i < 8) o0[i] = (__bf16)y; else o1[i - 8] = (__bf16)y;
  }
  __bf16* dst = oin + (size_t)row * 512 + lane * 16;
  *(bf16x8*)(dst)     = o0;
  *(bf16x8*)(dst + 8) = o1;
}

// ---------------------------------------------------------------------------
// Kernel 6: GEMM2  out = oin @ o_w + o_b + x  (residual). Wave tile 16x128,
// same [18 VMEM][8 WMMA] pipeline groups.
// ---------------------------------------------------------------------------
__global__ void __launch_bounds__(256)
k_gemm2(const __bf16* __restrict__ oin, const __bf16* __restrict__ owT,
        const float* __restrict__ ob, const float* __restrict__ x,
        float* __restrict__ out) {
  const int wid  = blockIdx.x * 8 + (threadIdx.x >> 5);
  const int mt   = wid & 511;        // 512 M tiles
  const int ns   = wid >> 9;         // 4 N strips of 128
  const int lane = threadIdx.x & 31;
  const int r = lane & 15, hi = lane >> 4;

  const __bf16* pa = oin + (size_t)(mt * 16 + r) * 512 + hi * 8;
  const __bf16* pb = owT + (size_t)(ns * 128 + r) * 512 + hi * 8;

  f32x8 acc[8] = {};
#pragma unroll
  for (int k0 = 0; k0 < 512; k0 += 32) {
    bf16x16 a = frag_a_at(pa, k0);
    bf16x16 bfr[8];
#pragma unroll
    for (int j = 0; j < 8; ++j) bfr[j] = frag_b_at(pb, j * 8192 + k0);
#pragma unroll
    for (int j = 0; j < 8; ++j) acc[j] = WMMA_BF16(a, bfr[j], acc[j]);
    __builtin_amdgcn_sched_group_barrier(SG_VMEMR, 18, 0);
    __builtin_amdgcn_sched_group_barrier(SG_MFMA,  8,  0);
  }
#pragma unroll
  for (int j = 0; j < 8; ++j) {
    const int c    = ns * 128 + j * 16 + r;
    const int rowb = mt * 16 + hi * 8;
    const float bias = ob[c];
    const float* px = x + (size_t)rowb * 512 + c;
    float* po = out + (size_t)rowb * 512 + c;
#pragma unroll
    for (int e = 0; e < 8; ++e) po[e * 512] = acc[j][e] + bias + px[e * 512];
  }
}

// ---------------------------------------------------------------------------
// Launch
// ---------------------------------------------------------------------------
extern "C" void kernel_launch(void* const* d_in, const int* in_sizes, int n_in,
                              void* d_out, int out_size, void* d_ws, size_t ws_size,
                              hipStream_t stream) {
  (void)in_sizes; (void)n_in; (void)out_size; (void)ws_size;
  const float* x    = (const float*)d_in[0];
  const float* uvqk = (const float*)d_in[1];
  const float* o_w  = (const float*)d_in[2];
  const float* o_b  = (const float*)d_in[3];
  const float* mask = (const float*)d_in[4];
  // d_in[5] x_offsets: regular (arange*N) layout -> b = t>>10, pos = t&1023.
  float* out = (float*)d_out;

  char* ws = (char*)d_ws;
  const size_t MB = 1024 * 1024;
  __bf16* xn   = (__bf16*)(ws + 0);          //  8 MB  normed x, bf16
  __bf16* qb   = (__bf16*)(ws + 8  * MB);    //  8 MB
  __bf16* kb   = (__bf16*)(ws + 16 * MB);    //  8 MB
  __bf16* vT   = (__bf16*)(ws + 24 * MB);    //  8 MB  V transposed per (b,h)
  float*  u_f  = (float*) (ws + 32 * MB);    // 16 MB
  float*  attn = (float*) (ws + 48 * MB);    // 16 MB
  __bf16* oin  = (__bf16*)(ws + 64 * MB);    //  8 MB
  __bf16* wT   = (__bf16*)(ws + 72 * MB);    //  2 MB  uvqk^T bf16
  __bf16* owT  = (__bf16*)(ws + 74 * MB + 512 * 1024); // 0.5 MB o_w^T bf16

  k_ln_x   <<<1024, 256, 0, stream>>>(x, xn);
  k_transcvt<<<(512 * 2048 + 255) / 256, 256, 0, stream>>>(uvqk, wT, 512, 2048);
  k_transcvt<<<(512 * 512  + 255) / 256, 256, 0, stream>>>(o_w, owT, 512, 512);
  k_gemm1  <<<1024, 256, 0, stream>>>(xn, wT, u_f, qb, kb, vT);
  k_attn   <<<512,  256, 0, stream>>>(qb, kb, vT, mask, attn);
  k_ln_gate<<<1024, 256, 0, stream>>>(attn, u_f, oin);
  k_gemm2  <<<256,  256, 0, stream>>>(oin, owT, o_b, x, out);
}
